// model_2379411882535
// MI455X (gfx1250) — matrix operand order repaired and checked by reading, not compiled, not run on hardware
//
#include <hip/hip_runtime.h>
#include <math.h>

typedef __attribute__((ext_vector_type(16))) _Float16 v16h;
typedef __attribute__((ext_vector_type(8)))  _Float16 v8h;
typedef __attribute__((ext_vector_type(8)))  float    v8f;

// ---------------------------------------------------------------------------
// WMMA GEMM on pre-converted f16 operands.
//   A16 : M x Kpad   (row-major, f16, zero-padded K, Kpad % 32 == 0)
//   Wt16: Npad x Kpad (f16, transposed weight: Wt16[n][k] = W[k][n], padded)
//   C   : M x Ncol   (f32, + bias, optional relu)
// One wave per 16x16 tile. Per K-chunk: 4x global_load_b128 + 1x v_wmma.
// ---------------------------------------------------------------------------
__global__ __launch_bounds__(32)
void wmma_gemm_kernel(const _Float16* __restrict__ A16,
                      const _Float16* __restrict__ Wt16,
                      const float* __restrict__ bias, float* __restrict__ C,
                      int Kpad, int Ncol, int relu_flag)
{
    const int lane = threadIdx.x;
    const int row0 = blockIdx.y * 16, col0 = blockIdx.x * 16;
    const int m = lane & 15, half = lane >> 4;
    // A fragment: lane(m,half) needs k = half*8+e (e<8) and 16+half*8+e (e>=8)
    const _Float16* arow = A16 + (size_t)(row0 + m) * Kpad + half * 8;
    // B fragment: lane(n=m,half) needs k = half*8+e (e<8) and 16+half*8+e (e>=8)
    const _Float16* brow = Wt16 + (size_t)(col0 + m) * Kpad + half * 8;
    v8f c = {};
    const int kch = Kpad >> 5;
    for (int kc = 0; kc < kch; ++kc) {
        const int k0 = kc * 32;
        v8h a0 = *(const v8h*)(arow + k0);
        v8h a1 = *(const v8h*)(arow + k0 + 16);
        v8h b0 = *(const v8h*)(brow + k0);
        v8h b1 = *(const v8h*)(brow + k0 + 16);
        v16h a = __builtin_shufflevector(a0, a1, 0,1,2,3,4,5,6,7,8,9,10,11,12,13,14,15);
        v16h b = __builtin_shufflevector(b0, b1, 0,1,2,3,4,5,6,7,8,9,10,11,12,13,14,15);
        c = __builtin_amdgcn_wmma_f32_16x16x32_f16(false, a, false, b,
                                                   (short)0, c, false, false);
    }
    const int gc = col0 + m;            // column for this lane
    if (gc < Ncol) {
        float bv = bias ? bias[gc] : 0.0f;
#pragma unroll
        for (int vg = 0; vg < 8; ++vg) {        // C layout: m = vg + 8*half
            int gr = row0 + vg + 8 * half;      // M is always a multiple of 16
            float o = c[vg] + bv;
            if (relu_flag) o = fmaxf(o, 0.0f);
            C[(size_t)gr * Ncol + gc] = o;
        }
    }
}

// f32 -> f16 with zero K-padding:  dst[r][c] (r<rows, c<colspad)
__global__ void cvt_pad_kernel(const float* __restrict__ src, _Float16* __restrict__ dst,
                               int rows, int cols, int colspad) {
    int idx = blockIdx.x * 256 + threadIdx.x;
    if (idx >= rows * colspad) return;
    int r = idx / colspad, c = idx - r * colspad;
    dst[idx] = (c < cols) ? (_Float16)src[(size_t)r * cols + c] : (_Float16)0.0f;
}
// weight W[K][Ncol] -> f16 transposed padded Wt[Npad][Kpad]
__global__ void cvt_pad_t_kernel(const float* __restrict__ w, _Float16* __restrict__ wt,
                                 int K, int Ncol, int Kpad, int Npad) {
    int idx = blockIdx.x * 256 + threadIdx.x;
    if (idx >= Npad * Kpad) return;
    int n = idx / Kpad, k = idx - n * Kpad;
    wt[idx] = (k < K && n < Ncol) ? (_Float16)w[(size_t)k * Ncol + n] : (_Float16)0.0f;
}

// ---------------------------------------------------------------------------
// small helpers
// ---------------------------------------------------------------------------
__device__ inline float leaky02(float x) { return x > 0.f ? x : 0.2f * x; }
__device__ inline float sigm(float x)    { return 1.0f / (1.0f + __expf(-x)); }
__device__ inline void atomicMaxF(float* a, float v) {
    if (v >= 0.f) atomicMax((int*)a, __float_as_int(v));
    else          atomicMin((unsigned int*)a, __float_as_uint(v));
}

__global__ void fill_kernel(float* p, float v, int n) {
    int i = blockIdx.x * 256 + threadIdx.x;
    if (i < n) p[i] = v;
}
__global__ void copy_kernel(float* d, const float* s, int n) {
    int i = blockIdx.x * 256 + threadIdx.x;
    if (i < n) d[i] = s[i];
}
__global__ void pad_copy_kernel(const float* x, float* hg, int n, int cin, int h) {
    int idx = blockIdx.x * 256 + threadIdx.x;
    if (idx >= n * h) return;
    int i = idx / h, c = idx - i * h;
    hg[idx] = (c < cin) ? x[(size_t)i * cin + c] : 0.0f;
}
__global__ void addrelu_kernel(float* acc, const float* t, const float* bias, int n, int h) {
    int idx = blockIdx.x * 256 + threadIdx.x;
    if (idx >= n * h) return;
    int c = idx % h;
    float o = t[idx] + (bias ? bias[c] : 0.0f);
    acc[idx] += fmaxf(o, 0.0f);
}

// ---------------------------------------------------------------------------
// degree / symmetric-norm coefficients
// ---------------------------------------------------------------------------
__global__ void deg_count_kernel(const int* dst, float* deg, int e) {
    int i = blockIdx.x * 256 + threadIdx.x;
    if (i < e) atomicAdd(&deg[dst[i]], 1.0f);
}
__global__ void dinv_kernel(float* deg_dinv, int n) {
    int i = blockIdx.x * 256 + threadIdx.x;
    if (i < n) { float d = deg_dinv[i] + 1.0f; deg_dinv[i] = rsqrtf(fmaxf(d, 1.0f)); }
}
__global__ void coef_kernel(const int* src, const int* dst, const float* dinv,
                            float* coef, int e) {
    int i = blockIdx.x * 256 + threadIdx.x;
    if (i < e) coef[i] = dinv[src[i]] * dinv[dst[i]];
}

// out[i] = scale*dinv[i]^2*h[i] (+ alpha*h0[i]); one block per node
__global__ void prop_init_kernel(float* out, const float* h, const float* h0,
                                 const float* dinv, float scale, float alpha, int hc) {
    int i = blockIdx.x;
    float d2 = dinv[i] * dinv[i];
    for (int c = threadIdx.x; c < hc; c += blockDim.x) {
        float o = scale * d2 * h[(size_t)i * hc + c];
        if (h0) o += alpha * h0[(size_t)i * hc + c];
        out[(size_t)i * hc + c] = o;
    }
}
// out[dst] += scale*(coef?coef[e]:1)*h[src]; one block per edge
__global__ void edge_scatter_kernel(float* out, const float* h, const int* src,
                                    const int* dst, const float* coef, float scale, int hc) {
    int e = blockIdx.x;
    int s = src[e], d = dst[e];
    float w = scale * (coef ? coef[e] : 1.0f);
    for (int c = threadIdx.x; c < hc; c += blockDim.x)
        atomicAdd(&out[(size_t)d * hc + c], w * h[(size_t)s * hc + c]);
}

// ---------------------------------------------------------------------------
// GAT pieces
// ---------------------------------------------------------------------------
__global__ void gemv_kernel(const float* h, const float* a, float* s, int n, int hc) {
    int i = blockIdx.x * 256 + threadIdx.x;
    if (i >= n) return;
    float acc = 0.f;
    for (int c = 0; c < hc; ++c) acc += h[(size_t)i * hc + c] * a[c];
    s[i] = acc;
}
__global__ void gat_selfmax_kernel(const float* ss, const float* sd, float* smax, int n) {
    int i = blockIdx.x * 256 + threadIdx.x;
    if (i < n) smax[i] = leaky02(ss[i] + sd[i]);
}
__global__ void gat_edgemax_kernel(const int* src, const int* dst, const float* ss,
                                   const float* sd, float* smax, int e) {
    int i = blockIdx.x * 256 + threadIdx.x;
    if (i >= e) return;
    atomicMaxF(&smax[dst[i]], leaky02(ss[src[i]] + sd[dst[i]]));
}
__global__ void gat_selfden_kernel(const float* ss, const float* sd, const float* smax,
                                   float* den, int n) {
    int i = blockIdx.x * 256 + threadIdx.x;
    if (i < n) den[i] = __expf(leaky02(ss[i] + sd[i]) - smax[i]);
}
__global__ void gat_edgeden_kernel(const int* src, const int* dst, const float* ss,
                                   const float* sd, const float* smax, float* eex,
                                   float* den, int e) {
    int i = blockIdx.x * 256 + threadIdx.x;
    if (i >= e) return;
    int s = src[i], d = dst[i];
    float ex = __expf(leaky02(ss[s] + sd[d]) - smax[d]);
    eex[i] = ex;
    atomicAdd(&den[d], ex);
}
__global__ void gat_agg_init_kernel(float* out, const float* hlin, const float* ss,
                                    const float* sd, const float* smax, const float* den,
                                    int hc) {
    int i = blockIdx.x;
    float w = __expf(leaky02(ss[i] + sd[i]) - smax[i]) / den[i];
    for (int c = threadIdx.x; c < hc; c += blockDim.x)
        out[(size_t)i * hc + c] = w * hlin[(size_t)i * hc + c];
}
__global__ void gat_agg_edge_kernel(float* out, const float* hlin, const int* src,
                                    const int* dst, const float* eex, const float* den,
                                    int hc) {
    int e = blockIdx.x;
    int s = src[e], d = dst[e];
    float w = eex[e] / den[d];
    for (int c = threadIdx.x; c < hc; c += blockDim.x)
        atomicAdd(&out[(size_t)d * hc + c], w * hlin[(size_t)s * hc + c]);
}

// ---------------------------------------------------------------------------
// GRU elementwise (GatedGraphConv)
// ---------------------------------------------------------------------------
__global__ void gru_kernel(float* hg, const float* gi, const float* gh, int n, int hc) {
    int idx = blockIdx.x * 256 + threadIdx.x;
    if (idx >= n * hc) return;
    int i = idx / hc, c = idx - i * hc;
    size_t b3 = (size_t)i * 3 * hc;
    float ir = gi[b3 + c],        hr = gh[b3 + c];
    float iz = gi[b3 + hc + c],   hz = gh[b3 + hc + c];
    float in = gi[b3 + 2*hc + c], hn = gh[b3 + 2*hc + c];
    float r = sigm(ir + hr), z = sigm(iz + hz);
    float nn = tanhf(in + r * hn);
    hg[idx] = (1.0f - z) * nn + z * hg[idx];
}

// ---------------------------------------------------------------------------
// conv heads
// ---------------------------------------------------------------------------
__global__ void wide_build_kernel(const float* x, const float* h2, float* wide,
                                  int n, int cin, int h) {
    int wtot = cin + h;
    int idx = blockIdx.x * 256 + threadIdx.x;
    if (idx >= n * wtot) return;
    int i = idx / wtot, f = idx - i * wtot;
    wide[idx] = (f < cin) ? x[(size_t)i * cin + f] : h2[(size_t)i * h + (f - cin)];
}
__global__ void conv1d_relu_kernel(const float* in, const float* w, const float* bias,
                                   float* out, int bn, int ic, int lin, int oc, int kw,
                                   int lout) {
    int idx = blockIdx.x * 256 + threadIdx.x;
    int total = bn * oc * lout;
    if (idx >= total) return;
    int xo = idx % lout;
    int o  = (idx / lout) % oc;
    int b  = idx / (lout * oc);
    float acc = bias[o];
    for (int i = 0; i < ic; ++i)
        for (int t = 0; t < kw; ++t)
            acc += in[((size_t)b * ic + i) * lin + xo + t] * w[((size_t)o * ic + i) * kw + t];
    out[idx] = fmaxf(acc, 0.0f);
}
__global__ void maxpool_kernel(const float* in, float* out, int bn, int cc, int lin,
                               int k, int s, int lout) {
    int idx = blockIdx.x * 256 + threadIdx.x;
    int total = bn * cc * lout;
    if (idx >= total) return;
    int xo = idx % lout;
    int c  = (idx / lout) % cc;
    int b  = idx / (lout * cc);
    const float* base = in + ((size_t)b * cc + c) * lin + xo * s;
    float m = base[0];
    for (int t = 1; t < k; ++t) m = fmaxf(m, base[t]);
    out[idx] = m;
}
__global__ void final_kernel(const float* wf, const float* nf, const float* Ww,
                             const float* bwm, const float* Wn, const float* bnm,
                             float* out, int bn) {
    int b = blockIdx.x * 256 + threadIdx.x;
    if (b >= bn) return;
    float a = bwm[0];
    for (int j = 0; j < 38; ++j) a += wf[b * 38 + j] * Ww[j];
    float c = bnm[0];
    for (int j = 0; j < 24; ++j) c += nf[b * 24 + j] * Wn[j];
    out[b] = a + c;
}

// ---------------------------------------------------------------------------
// host orchestration
// ---------------------------------------------------------------------------
extern "C" void kernel_launch(void* const* d_in, const int* in_sizes, int n_in,
                              void* d_out, int out_size, void* d_ws, size_t ws_size,
                              hipStream_t stream) {
    (void)in_sizes; (void)n_in; (void)out_size; (void)ws_size;
    constexpr int Nn = 51200, Ee = 819200, CIN = 116, Hh = 200, Bb = 256;
    constexpr int KPMAX = 224, NPMAX = 608;   // max padded K / padded Ncol

    const float* x     = (const float*)d_in[0];
    const int*   esrc  = (const int*)d_in[1];
    const int*   edst  = (const int*)d_in[2];
    const float* W_gat = (const float*)d_in[3];
    const float* a_src = (const float*)d_in[4];
    const float* a_dst = (const float*)d_in[5];
    const float* b_gat = (const float*)d_in[6];
    const float* W_gcn = (const float*)d_in[7];
    const float* b_gcn = (const float*)d_in[8];
    const float* Wa1   = (const float*)d_in[9];
    const float* ba1   = (const float*)d_in[10];
    const float* Wa2   = (const float*)d_in[11];
    const float* ba2   = (const float*)d_in[12];
    const float* W_ggnn= (const float*)d_in[13];
    const float* W_ih  = (const float*)d_in[14];
    const float* W_hh  = (const float*)d_in[15];
    const float* b_ih  = (const float*)d_in[16];
    const float* b_hh  = (const float*)d_in[17];
    const float* W_gat2= (const float*)d_in[18];
    const float* a_s2  = (const float*)d_in[19];
    const float* a_d2  = (const float*)d_in[20];
    const float* b_gat2= (const float*)d_in[21];
    const float* W_gcn2= (const float*)d_in[22];
    const float* b_gcn2= (const float*)d_in[23];
    const float* Wa21  = (const float*)d_in[24];
    const float* ba21  = (const float*)d_in[25];
    const float* Wa22  = (const float*)d_in[26];
    const float* ba22  = (const float*)d_in[27];
    const float* cw1   = (const float*)d_in[28];
    const float* bw1   = (const float*)d_in[29];
    const float* cw2   = (const float*)d_in[30];
    const float* bw2   = (const float*)d_in[31];
    const float* cw3   = (const float*)d_in[32];
    const float* bw3   = (const float*)d_in[33];
    const float* Ww    = (const float*)d_in[34];
    const float* bwm   = (const float*)d_in[35];
    const float* cn1   = (const float*)d_in[36];
    const float* bn1   = (const float*)d_in[37];
    const float* cn2   = (const float*)d_in[38];
    const float* bn2   = (const float*)d_in[39];
    const float* cn3   = (const float*)d_in[40];
    const float* bn3   = (const float*)d_in[41];
    const float* Wn    = (const float*)d_in[42];
    const float* bnm   = (const float*)d_in[43];
    float* out = (float*)d_out;

    // workspace carve-out (all sizes multiples of 4 floats -> 16B alignment)
    float* ws = (float*)d_ws;
    size_t off = 0;
    auto carve = [&](size_t n) { float* p = ws + off; off += n; return p; };
    float* dinv = carve(Nn);
    float* coef = carve(Ee);
    float* hg   = carve((size_t)Nn * Hh);
    float* h0   = carve((size_t)Nn * Hh);
    float* hA   = carve((size_t)Nn * Hh);
    float* hB   = carve((size_t)Nn * Hh);
    float* acc1 = carve((size_t)Nn * Hh);
    float* acc2 = carve((size_t)Nn * Hh);
    float* tA   = carve((size_t)Nn * Hh);
    float* tB   = carve((size_t)Nn * Hh);
    float* gi   = carve((size_t)Nn * 3 * Hh);
    float* gh   = carve((size_t)Nn * 3 * Hh);
    float* ssrc = carve(Nn);
    float* sdst = carve(Nn);
    float* smax = carve(Nn);
    float* sden = carve(Nn);
    float* eex  = carve(Ee);
    float* wide = carve((size_t)Nn * (CIN + Hh));
    float* c1b  = carve((size_t)Bb * 100 * 316);
    float* p1b  = carve((size_t)Bb * 100 * 156);
    float* c2b  = carve((size_t)Bb * 50 * 156);
    float* p2b  = carve((size_t)Bb * 50 * 78);
    float* c3b  = carve((size_t)Bb * 1 * 76);
    float* featW = carve((size_t)Bb * 38);
    float* featN = carve((size_t)Bb * 24);
    _Float16* a16  = (_Float16*)carve((size_t)Nn * KPMAX / 2);   // f16 activations
    _Float16* w16t = (_Float16*)carve((size_t)NPMAX * KPMAX / 2);// f16 weight^T

    auto g = [](int n) { return (n + 255) / 256; };
    // convert activations A[Nn x K] -> a16 (zero-padded to Kpad)
    auto prepA = [&](const float* A, int K) {
        int Kp = (K + 31) & ~31;
        cvt_pad_kernel<<<g(Nn * Kp), 256, 0, stream>>>(A, a16, Nn, K, Kp);
        return Kp;
    };
    // convert weight + run GEMM on already-prepared a16
    auto gemmW = [&](int Kp, int K, const float* W, const float* bias, float* C,
                     int Ncol, int relu) {
        int Np = (Ncol + 15) & ~15;
        cvt_pad_t_kernel<<<g(Np * Kp), 256, 0, stream>>>(W, w16t, K, Ncol, Kp, Np);
        dim3 grid(Np / 16, Nn / 16);
        wmma_gemm_kernel<<<grid, dim3(32), 0, stream>>>(a16, w16t, bias, C, Kp, Ncol, relu);
    };
    auto gemm = [&](const float* A, const float* W, const float* bias, float* C,
                    int K, int Ncol, int relu) {
        int Kp = prepA(A, K);
        gemmW(Kp, K, W, bias, C, Ncol, relu);
    };

    // ---- degrees & normalization coefficients ----
    fill_kernel<<<g(Nn), 256, 0, stream>>>(dinv, 0.0f, Nn);
    deg_count_kernel<<<g(Ee), 256, 0, stream>>>(edst, dinv, Ee);
    dinv_kernel<<<g(Nn), 256, 0, stream>>>(dinv, Nn);
    coef_kernel<<<g(Ee), 256, 0, stream>>>(esrc, edst, dinv, coef, Ee);

    // ---- GatedGraphConv (3 layers, no self loops) ----
    pad_copy_kernel<<<g(Nn * Hh), 256, 0, stream>>>(x, hg, Nn, CIN, Hh);
    for (int l = 0; l < 3; ++l) {
        int Kp = prepA(hg, Hh);                                  // convert hg once
        gemmW(Kp, Hh, W_ggnn + (size_t)l * Hh * Hh, nullptr, tA, Hh, 0);
        gemmW(Kp, Hh, W_hh, b_hh, gh, 3 * Hh, 0);                // gh from same hg
        fill_kernel<<<g(Nn * Hh), 256, 0, stream>>>(tB, 0.0f, Nn * Hh);
        edge_scatter_kernel<<<Ee, 256, 0, stream>>>(tB, tA, esrc, edst, nullptr, 1.0f, Hh);
        gemm(tB, W_ih, b_ih, gi, Hh, 3 * Hh, 0);
        gru_kernel<<<g(Nn * Hh), 256, 0, stream>>>(hg, gi, gh, Nn, Hh);
    }

    auto gat = [&](const float* hin, int K, const float* W, const float* avs,
                   const float* avd, const float* bias, float* accbuf) {
        gemm(hin, W, nullptr, tA, K, Hh, 0);
        gemv_kernel<<<g(Nn), 256, 0, stream>>>(tA, avs, ssrc, Nn, Hh);
        gemv_kernel<<<g(Nn), 256, 0, stream>>>(tA, avd, sdst, Nn, Hh);
        gat_selfmax_kernel<<<g(Nn), 256, 0, stream>>>(ssrc, sdst, smax, Nn);
        gat_edgemax_kernel<<<g(Ee), 256, 0, stream>>>(esrc, edst, ssrc, sdst, smax, Ee);
        gat_selfden_kernel<<<g(Nn), 256, 0, stream>>>(ssrc, sdst, smax, sden, Nn);
        gat_edgeden_kernel<<<g(Ee), 256, 0, stream>>>(esrc, edst, ssrc, sdst, smax, eex, sden, Ee);
        gat_agg_init_kernel<<<Nn, 256, 0, stream>>>(tB, tA, ssrc, sdst, smax, sden, Hh);
        gat_agg_edge_kernel<<<Ee, 256, 0, stream>>>(tB, tA, esrc, edst, eex, sden, Hh);
        addrelu_kernel<<<g(Nn * Hh), 256, 0, stream>>>(accbuf, tB, bias, Nn, Hh);
    };
    auto gcn = [&](const float* hin, int K, const float* W, const float* bias, float* accbuf) {
        gemm(hin, W, nullptr, tA, K, Hh, 0);
        prop_init_kernel<<<Nn, 256, 0, stream>>>(tB, tA, nullptr, dinv, 1.0f, 0.0f, Hh);
        edge_scatter_kernel<<<Ee, 256, 0, stream>>>(tB, tA, esrc, edst, coef, 1.0f, Hh);
        addrelu_kernel<<<g(Nn * Hh), 256, 0, stream>>>(accbuf, tB, bias, Nn, Hh);
    };
    auto appnp = [&](const float* hin, int K, int Kmid, const float* W1, const float* b1,
                     const float* W2, const float* b2, float* accbuf) {
        gemm(hin, W1, b1, tA, K, Kmid, 1);           // relu(h @ W1 + b1)
        gemm(tA, W2, b2, h0, Kmid, Hh, 0);           // h0
        copy_kernel<<<g(Nn * Hh), 256, 0, stream>>>(hA, h0, Nn * Hh);
        float* ha = hA; float* hb = hB;
        for (int it = 0; it < 10; ++it) {            // h = 0.8*prop(h) + 0.2*h0
            prop_init_kernel<<<Nn, 256, 0, stream>>>(hb, ha, h0, dinv, 0.8f, 0.2f, Hh);
            edge_scatter_kernel<<<Ee, 256, 0, stream>>>(hb, ha, esrc, edst, coef, 0.8f, Hh);
            float* t = ha; ha = hb; hb = t;
        }
        addrelu_kernel<<<g(Nn * Hh), 256, 0, stream>>>(accbuf, ha, nullptr, Nn, Hh);
    };

    // ---- block 1: h1 = relu(gat)+relu(gcn)+relu(appnp)+hg ----
    copy_kernel<<<g(Nn * Hh), 256, 0, stream>>>(acc1, hg, Nn * Hh);
    gat(x, CIN, W_gat, a_src, a_dst, b_gat, acc1);
    gcn(x, CIN, W_gcn, b_gcn, acc1);
    appnp(x, CIN, 116, Wa1, ba1, Wa2, ba2, acc1);

    // ---- block 2: h2 = relu(gat2)+relu(gcn2)+relu(appnp2) ----
    fill_kernel<<<g(Nn * Hh), 256, 0, stream>>>(acc2, 0.0f, Nn * Hh);
    gat(acc1, Hh, W_gat2, a_s2, a_d2, b_gat2, acc2);
    gcn(acc1, Hh, W_gcn2, b_gcn2, acc2);
    appnp(acc1, Hh, Hh, Wa21, ba21, Wa22, ba22, acc2);

    // ---- conv heads ----
    wide_build_kernel<<<g(Nn * (CIN + Hh)), 256, 0, stream>>>(x, acc2, wide, Nn, CIN, Hh);
    auto head = [&](const float* tin, int L0, const float* w1, const float* bi1,
                    const float* w2, const float* bi2, const float* w3, const float* bi3,
                    float* feat) {
        int L1 = L0 - 2;
        int P1 = (L1 - 3) / 2 + 1;
        int P2 = (P1 - 2) / 2 + 1;
        int L3 = P2 - 2;
        int P3 = (L3 - 1) / 2 + 1;
        conv1d_relu_kernel<<<g(Bb * 100 * L1), 256, 0, stream>>>(tin, w1, bi1, c1b, Bb, 200, L0, 100, 3, L1);
        maxpool_kernel<<<g(Bb * 100 * P1), 256, 0, stream>>>(c1b, p1b, Bb, 100, L1, 3, 2, P1);
        conv1d_relu_kernel<<<g(Bb * 50 * P1), 256, 0, stream>>>(p1b, w2, bi2, c2b, Bb, 100, P1, 50, 1, P1);
        maxpool_kernel<<<g(Bb * 50 * P2), 256, 0, stream>>>(c2b, p2b, Bb, 50, P1, 2, 2, P2);
        conv1d_relu_kernel<<<g(Bb * 1 * L3), 256, 0, stream>>>(p2b, w3, bi3, c3b, Bb, 50, P2, 1, 3, L3);
        maxpool_kernel<<<g(Bb * 1 * P3), 256, 0, stream>>>(c3b, feat, Bb, 1, L3, 1, 2, P3);
    };
    head(wide, CIN + Hh, cw1, bw1, cw2, bw2, cw3, bw3, featW); // -> (B,38)
    head(acc2, Hh,       cn1, bn1, cn2, bn2, cn3, bn3, featN); // -> (B,24)

    final_kernel<<<g(Bb), 256, 0, stream>>>(featW, featN, Ww, bwm, Wn, bnm, out, Bb);
}
